// SpatialGNN_12463995093933
// MI455X (gfx1250) — matrix operand order repaired and checked by reading, not compiled, not run on hardware
//
#include <hip/hip_runtime.h>
#include <stdint.h>
#include <math.h>

#define TPB 256

typedef __bf16 bf16_t;
typedef bf16_t v16bf __attribute__((ext_vector_type(16)));
typedef float  v8f   __attribute__((ext_vector_type(8)));

__device__ __forceinline__ unsigned short f32_to_bf16_raw(float f) {
  union { float f; unsigned int u; } x; x.f = f;
  unsigned int r = 0x7FFFu + ((x.u >> 16) & 1u);   // round-to-nearest-even
  return (unsigned short)((x.u + r) >> 16);
}

// ---------------------------------------------------------------------------
// Graph-norm helpers
// ---------------------------------------------------------------------------
__global__ void gnn_zero_f32(float* __restrict__ p, int n) {
  int i = blockIdx.x * blockDim.x + threadIdx.x;
  if (i < n) p[i] = 0.0f;
}

__global__ void gnn_degree(const int* __restrict__ dstv, float* __restrict__ deg, int E) {
  int i = blockIdx.x * blockDim.x + threadIdx.x;
  if (i < E) atomicAdd(&deg[dstv[i]], 1.0f);
}

__global__ void gnn_rsqrt_inplace(float* __restrict__ deg, int n) {
  int i = blockIdx.x * blockDim.x + threadIdx.x;
  if (i < n) deg[i] = rsqrtf(deg[i] + 1.0f);   // dis = rsqrt(deg + 1 self loop)
}

// ---------------------------------------------------------------------------
// Pack fp32 -> bf16 WMMA fragments
// A fragment (16x32, MxK): lane L holds row M=L&15; lanes 0-15 carry K offsets
// {0..7,16..23}, lanes 16-31 carry {8..15,24..31}; 2 K-values per VGPR.
// Element e of the v16bf: v=e>>1, p=e&1 -> klocal.
// ---------------------------------------------------------------------------
__global__ void gnn_pack_a(const float* __restrict__ X, unsigned short* __restrict__ Apk,
                           int mtiles, int ktiles, int K) {
  int idx = blockIdx.x * blockDim.x + threadIdx.x;
  int total = mtiles * ktiles * 32;
  if (idx >= total) return;
  int lane = idx & 31;
  int frag = idx >> 5;
  int kt = frag % ktiles;
  int mt = frag / ktiles;
  int m  = mt * 16 + (lane & 15);
  int kb = (lane < 16) ? 0 : 8;
  unsigned short* dst = Apk + (size_t)frag * 512 + lane * 16;
  const float* row = X + (size_t)m * K;
#pragma unroll
  for (int e = 0; e < 16; ++e) {
    int v = e >> 1, p = e & 1;
    int kl = (v < 4) ? (kb + 2 * v + p) : (kb + 16 + 2 * (v - 4) + p);
    int k  = kt * 32 + kl;
    float f = (k < K) ? row[k] : 0.0f;
    dst[e] = f32_to_bf16_raw(f);
  }
}

// B fragment (32x16, KxN): lane L holds column N=L&15; lanes 0-15 carry K offsets
// {0..7,16..23}, lanes 16-31 carry {8..15,24..31} (same order as A).
__global__ void gnn_pack_w(const float* __restrict__ W, unsigned short* __restrict__ Wpk,
                           int ktiles, int ntiles, int K, int Ncol) {
  int idx = blockIdx.x * blockDim.x + threadIdx.x;
  int total = ktiles * ntiles * 32;
  if (idx >= total) return;
  int lane = idx & 31;
  int frag = idx >> 5;
  int nt = frag % ntiles;
  int kt = frag / ntiles;
  int n  = nt * 16 + (lane & 15);
  int kb = (lane >> 4) * 8;
  unsigned short* dst = Wpk + (size_t)frag * 512 + lane * 16;
#pragma unroll
  for (int e = 0; e < 16; ++e) {
    int k = kt * 32 + kb + ((e < 8) ? e : (8 + e));
    float f = (k < K) ? W[(size_t)k * Ncol + n] : 0.0f;
    dst[e] = f32_to_bf16_raw(f);
  }
}

// ---------------------------------------------------------------------------
// WMMA GEMM: H[N, C] = A[N, K] @ W[K, C], bf16 in, f32 accumulate.
// One wave computes a 16x64 output strip (4 accumulators, A-fragment reuse x4).
// ---------------------------------------------------------------------------
__global__ void __launch_bounds__(TPB)
gnn_gemm_wmma(const unsigned short* __restrict__ Apk,
              const unsigned short* __restrict__ Wpk,
              float* __restrict__ H, int mtiles, int ntiles, int ktiles) {
  const int wave = blockIdx.x * (TPB >> 5) + (threadIdx.x >> 5);
  const int lane = threadIdx.x & 31;
  const int ng   = ntiles >> 2;                 // all ntiles are multiples of 4
  if (wave >= mtiles * ng) return;              // wave-uniform exit: EXEC stays all-1
  const int mt  = wave / ng;
  const int nt0 = (wave - mt * ng) << 2;

  const v16bf* A = (const v16bf*)Apk + (size_t)mt * ktiles * 32 + lane;
  const v16bf* B = (const v16bf*)Wpk + (size_t)nt0 * 32 + lane;

  v8f acc0 = {}, acc1 = {}, acc2 = {}, acc3 = {};
  for (int kt = 0; kt < ktiles; ++kt) {
    v16bf a = A[(size_t)kt * 32];
    const v16bf* Bk = B + (size_t)kt * ntiles * 32;
    v16bf b0 = Bk[0];
    v16bf b1 = Bk[32];
    v16bf b2 = Bk[64];
    v16bf b3 = Bk[96];
    if (kt + 1 < ktiles) __builtin_prefetch(&A[(size_t)(kt + 1) * 32], 0, 1);
    acc0 = __builtin_amdgcn_wmma_f32_16x16x32_bf16(false, a, false, b0, (short)0, acc0, false, false);
    acc1 = __builtin_amdgcn_wmma_f32_16x16x32_bf16(false, a, false, b1, (short)0, acc1, false, false);
    acc2 = __builtin_amdgcn_wmma_f32_16x16x32_bf16(false, a, false, b2, (short)0, acc2, false, false);
    acc3 = __builtin_amdgcn_wmma_f32_16x16x32_bf16(false, a, false, b3, (short)0, acc3, false, false);
  }

  // C/D layout: VGPR r -> row (mt*16 + r + 8*(lane>>4)), col = nt*16 + (lane&15)
  const int C = ntiles * 16;
  const int n = lane & 15;
  const int mbase = mt * 16 + ((lane >> 4) << 3);
  float* Hrow = H + (size_t)mbase * C + nt0 * 16 + n;
#pragma unroll
  for (int r = 0; r < 8; ++r) {
    float* p = Hrow + (size_t)r * C;
    p[0]  = acc0[r];
    p[16] = acc1[r];
    p[32] = acc2[r];
    p[48] = acc3[r];
  }
}

// ---------------------------------------------------------------------------
// GCN aggregation: out = h*dis^2 + bias, then atomic scatter of h[src]*norm_e
// ---------------------------------------------------------------------------
__global__ void gnn_self_init(const float* __restrict__ h, const float* __restrict__ dis,
                              const float* __restrict__ bias, float* __restrict__ out,
                              long long total, int C) {
  long long idx = (long long)blockIdx.x * blockDim.x + threadIdx.x;
  if (idx >= total) return;
  int i = (int)(idx / C);
  int c = (int)(idx - (long long)i * C);
  float d = dis[i];
  out[idx] = h[idx] * (d * d) + bias[c];
}

__global__ void gnn_edge_scatter(const int* __restrict__ srcv, const int* __restrict__ dstv,
                                 const float* __restrict__ dis, const float* __restrict__ h,
                                 float* __restrict__ out, int E, int C) {
  long long idx = (long long)blockIdx.x * blockDim.x + threadIdx.x;
  int g = C >> 2;
  long long total = (long long)E * g;
  if (idx >= total) return;
  int e  = (int)(idx / g);
  int c4 = (int)(idx - (long long)e * g) * 4;
  int s = srcv[e], d = dstv[e];
  float w = dis[s] * dis[d];
  const float4 hv = *(const float4*)(h + (size_t)s * C + c4);
  float* o = out + (size_t)d * C + c4;
  atomicAdd(o + 0, hv.x * w);
  atomicAdd(o + 1, hv.y * w);
  atomicAdd(o + 2, hv.z * w);
  atomicAdd(o + 3, hv.w * w);
}

__global__ void gnn_bn_act(float* __restrict__ xio, const float* __restrict__ g,
                           const float* __restrict__ be, const float* __restrict__ mu,
                           const float* __restrict__ va, long long total, int C, int relu) {
  long long idx = (long long)blockIdx.x * blockDim.x + threadIdx.x;
  if (idx >= total) return;
  int c = (int)(idx % C);
  float s = g[c] * rsqrtf(va[c] + 1e-5f);
  float v = (xio[idx] - mu[c]) * s + be[c];
  xio[idx] = relu ? fmaxf(v, 0.0f) : v;
}

// ---------------------------------------------------------------------------
// Graph pooling: pooled[f] = (mean + max)/2 over nodes, C = 64
// ---------------------------------------------------------------------------
__global__ void gnn_pool(const float* __restrict__ act, float* __restrict__ pooled, int n) {
  __shared__ float ssum[512];
  __shared__ float smax[512];
  int f = threadIdx.x & 63;
  int c = threadIdx.x >> 6;      // 8 chunks per feature with 512 threads
  float s = 0.0f, m = -INFINITY;
  for (int i = c; i < n; i += 8) {
    float v = act[(size_t)i * 64 + f];
    s += v;
    m = fmaxf(m, v);
  }
  ssum[threadIdx.x] = s;
  smax[threadIdx.x] = m;
  __syncthreads();
  if (c == 0) {
#pragma unroll
    for (int j = 1; j < 8; ++j) {
      s += ssum[f + 64 * j];
      m = fmaxf(m, smax[f + 64 * j]);
    }
    pooled[f] = (s / (float)n + m) * 0.5f;
  }
}

// ---------------------------------------------------------------------------
// Tiny MLP heads (64->32->16->{10,1}), BN+ReLU after first two linears
// ---------------------------------------------------------------------------
struct HeadParams {
  const float* W0; const float* b0;
  const float* W1; const float* b1;
  const float* W2; const float* b2;
  const float* g0; const float* be0; const float* mu0; const float* v0;
  const float* g1; const float* be1; const float* mu1; const float* v1;
};

__device__ void gnn_run_head(const float* in64, const HeadParams& p, float* out, int outdim) {
  float z1[32], z2[16];
  for (int j = 0; j < 32; ++j) {
    float a = p.b0[j];
    for (int i = 0; i < 64; ++i) a += in64[i] * p.W0[i * 32 + j];
    a = (a - p.mu0[j]) * (p.g0[j] * rsqrtf(p.v0[j] + 1e-5f)) + p.be0[j];
    z1[j] = fmaxf(a, 0.0f);
  }
  for (int j = 0; j < 16; ++j) {
    float a = p.b1[j];
    for (int i = 0; i < 32; ++i) a += z1[i] * p.W1[i * 16 + j];
    a = (a - p.mu1[j]) * (p.g1[j] * rsqrtf(p.v1[j] + 1e-5f)) + p.be1[j];
    z2[j] = fmaxf(a, 0.0f);
  }
  for (int j = 0; j < outdim; ++j) {
    float a = p.b2[j];
    for (int i = 0; i < 16; ++i) a += z2[i] * p.W2[i * outdim + j];
    out[j] = a;
  }
}

__global__ void gnn_heads(const float* __restrict__ pooled, HeadParams cls, HeadParams reg,
                          float* __restrict__ out) {
  if (threadIdx.x == 0) {
    float logits[10];
    gnn_run_head(pooled, cls, logits, 10);
    for (int j = 0; j < 10; ++j) out[j] = logits[j];
    float r;
    gnn_run_head(pooled, reg, &r, 1);
    out[10] = 1.0f / (1.0f + expf(-r));
  }
}

// ---------------------------------------------------------------------------
// Host orchestration
// ---------------------------------------------------------------------------
static inline unsigned int cdivll(long long a, long long b) { return (unsigned int)((a + b - 1) / b); }

extern "C" void kernel_launch(void* const* d_in, const int* in_sizes, int n_in,
                              void* d_out, int out_size, void* d_ws, size_t ws_size,
                              hipStream_t stream) {
  (void)n_in; (void)out_size; (void)ws_size;
  const int IN_DIM = 2000;
  const int N = in_sizes[0] / IN_DIM;
  const int E = in_sizes[1] / 2;

  const float* x   = (const float*)d_in[0];
  const int* ei    = (const int*)d_in[1];
  const int* srcv  = ei;
  const int* dstv  = ei + E;
  const float* Ws[4] = {(const float*)d_in[2], (const float*)d_in[3], (const float*)d_in[4], (const float*)d_in[5]};
  const float* bs[4] = {(const float*)d_in[6], (const float*)d_in[7], (const float*)d_in[8], (const float*)d_in[9]};
  // bn_params[i] = (gamma, beta, mean, var) at indices 10+4i .. 13+4i
  // cls_lin: 26..31, cls_bn: 32..39, reg_lin: 40..45, reg_bn: 46..53

  int dims[5] = {IN_DIM, 512, 256, 128, 64};

  // ---- workspace layout ----
  char* ws = (char*)d_ws;
  size_t off = 0;
  auto salloc = [&](size_t bytes) -> char* {
    char* p = ws + off;
    off += (bytes + 255) & ~(size_t)255;
    return p;
  };
  float* dis = (float*)salloc((size_t)N * 4);
  unsigned short* wpk[4];
  for (int L = 0; L < 4; ++L) {
    int kt = (dims[L] + 31) / 32, nt = dims[L + 1] / 16;
    wpk[L] = (unsigned short*)salloc((size_t)kt * nt * 512 * 2);
  }
  const int mtiles = N / 16;
  const int max_kt = (IN_DIM + 31) / 32;   // 63
  unsigned short* apk = (unsigned short*)salloc((size_t)mtiles * max_kt * 512 * 2);
  float* hbuf   = (float*)salloc((size_t)N * 512 * 4);
  float* act    = (float*)salloc((size_t)N * 512 * 4);
  float* pooled = (float*)salloc(64 * 4);

  // ---- degree / symmetric norm ----
  gnn_zero_f32<<<cdivll(N, TPB), TPB, 0, stream>>>(dis, N);
  gnn_degree<<<cdivll(E, TPB), TPB, 0, stream>>>(dstv, dis, E);
  gnn_rsqrt_inplace<<<cdivll(N, TPB), TPB, 0, stream>>>(dis, N);

  // ---- pack weights (bf16 WMMA B fragments; all fit in L2) ----
  for (int L = 0; L < 4; ++L) {
    int K = dims[L], C = dims[L + 1];
    int kt = (K + 31) / 32, nt = C / 16;
    gnn_pack_w<<<cdivll((long long)kt * nt * 32, TPB), TPB, 0, stream>>>(Ws[L], wpk[L], kt, nt, K, C);
  }

  // ---- 4 GCN layers ----
  const float* cur = x;
  for (int L = 0; L < 4; ++L) {
    int K = dims[L], C = dims[L + 1];
    int kt = (K + 31) / 32, nt = C / 16;
    long long nc = (long long)N * C;

    gnn_pack_a<<<cdivll((long long)mtiles * kt * 32, TPB), TPB, 0, stream>>>(cur, apk, mtiles, kt, K);

    long long waves = (long long)mtiles * (nt >> 2);
    gnn_gemm_wmma<<<cdivll(waves, TPB / 32), TPB, 0, stream>>>(apk, wpk[L], hbuf, mtiles, nt, kt);

    gnn_self_init<<<cdivll(nc, TPB), TPB, 0, stream>>>(hbuf, dis, bs[L], act, nc, C);
    gnn_edge_scatter<<<cdivll((long long)E * (C >> 2), TPB), TPB, 0, stream>>>(srcv, dstv, dis, hbuf, act, E, C);

    const float* g  = (const float*)d_in[10 + 4 * L];
    const float* be = (const float*)d_in[11 + 4 * L];
    const float* mu = (const float*)d_in[12 + 4 * L];
    const float* va = (const float*)d_in[13 + 4 * L];
    gnn_bn_act<<<cdivll(nc, TPB), TPB, 0, stream>>>(act, g, be, mu, va, nc, C, (L < 3) ? 1 : 0);
    cur = act;
  }

  // ---- pooling ----
  gnn_pool<<<1, 512, 0, stream>>>(act, pooled, N);

  // ---- heads ----
  HeadParams cls, reg;
  cls.W0 = (const float*)d_in[26]; cls.b0 = (const float*)d_in[27];
  cls.W1 = (const float*)d_in[28]; cls.b1 = (const float*)d_in[29];
  cls.W2 = (const float*)d_in[30]; cls.b2 = (const float*)d_in[31];
  cls.g0 = (const float*)d_in[32]; cls.be0 = (const float*)d_in[33];
  cls.mu0 = (const float*)d_in[34]; cls.v0 = (const float*)d_in[35];
  cls.g1 = (const float*)d_in[36]; cls.be1 = (const float*)d_in[37];
  cls.mu1 = (const float*)d_in[38]; cls.v1 = (const float*)d_in[39];
  reg.W0 = (const float*)d_in[40]; reg.b0 = (const float*)d_in[41];
  reg.W1 = (const float*)d_in[42]; reg.b1 = (const float*)d_in[43];
  reg.W2 = (const float*)d_in[44]; reg.b2 = (const float*)d_in[45];
  reg.g0 = (const float*)d_in[46]; reg.be0 = (const float*)d_in[47];
  reg.mu0 = (const float*)d_in[48]; reg.v0 = (const float*)d_in[49];
  reg.g1 = (const float*)d_in[50]; reg.be1 = (const float*)d_in[51];
  reg.mu1 = (const float*)d_in[52]; reg.v1 = (const float*)d_in[53];

  gnn_heads<<<1, 32, 0, stream>>>(pooled, cls, reg, (float*)d_out);
}
